// FalconH1AttentionDecoderLayer_14070312861953
// MI455X (gfx1250) — matrix operand order repaired and checked by reading, not compiled, not run on hardware
//
#include <hip/hip_runtime.h>
#include <hip/hip_bf16.h>

typedef __attribute__((ext_vector_type(16))) _Float16 v16h;
typedef __attribute__((ext_vector_type(8)))  _Float16 v8h;
typedef __attribute__((ext_vector_type(8)))  float    v8f;

// ---------------- problem constants ----------------
static constexpr int HIDDEN   = 4096;
static constexpr int N_HEADS  = 32;
static constexpr int N_KVH    = 8;
static constexpr int HD       = 128;
static constexpr int Q_SIZE   = N_HEADS * HD;       // 4096
static constexpr int KV_SIZE  = N_KVH * HD;         // 1024
static constexpr int QKV_N    = Q_SIZE + 2 * KV_SIZE; // 6144
static constexpr int SEQ      = 2048;
static constexpr float KEY_MULT = 0.5f;
static constexpr float ATTN_SCALE = 0.08838834764831845f;   // 128^-0.5
static constexpr float NEG_LN_THETA_OVER_64 = -0.39575681279f; // -ln(1e11)/64

// ---------------- WMMA fragment loaders ----------------
// A-matrix 16x32 f16 layout (ISA 7.12.2): lane&15 = M row;
// half h: K = (h<8 ? h : h+8) + 8*(lane>=16). Two contiguous 16B chunks.
__device__ __forceinline__ v16h load_fragA(const _Float16* __restrict__ base,
                                           int row0, int stride, int k0, int lane) {
    const int hi = lane >> 4;
    const _Float16* p = base + (size_t)(row0 + (lane & 15)) * stride + k0 + hi * 8;
    v8h lo = *(const v8h*)(p);
    v8h hh = *(const v8h*)(p + 16);
    v16h r;
#pragma unroll
    for (int i = 0; i < 8; ++i) { r[i] = lo[i]; r[i + 8] = hh[i]; }
    return r;
}

// B-matrix 32x16 f16 (loaded from B^T, row-major [N][K]): lane&15 = N col;
// half h: K = (h<8 ? h : h+8) + 8*(lane>=16). Two contiguous 16B chunks.
__device__ __forceinline__ v16h load_fragB(const _Float16* __restrict__ base,
                                           int n0, int stride, int k0, int lane) {
    const _Float16* p = base + (size_t)(n0 + (lane & 15)) * stride + k0 + ((lane >> 4) << 3);
    v8h lo = *(const v8h*)(p);
    v8h hh = *(const v8h*)(p + 16);
    v16h r;
#pragma unroll
    for (int i = 0; i < 8; ++i) { r[i] = lo[i]; r[i + 8] = hh[i]; }
    return r;
}

__device__ __forceinline__ v8f wmma_f16(v16h a, v16h b, v8f c) {
    return __builtin_amdgcn_wmma_f32_16x16x32_f16(false, a, false, b, (short)0, c, false, false);
}

// ---------------- conversion kernels ----------------
__global__ void cvt_f32_to_f16(const float* __restrict__ src, _Float16* __restrict__ dst, int n) {
    for (int i = blockIdx.x * blockDim.x + threadIdx.x; i < n; i += gridDim.x * blockDim.x)
        dst[i] = (_Float16)src[i];
}

// dst[cols][rows] = (f16) src[rows][cols]   (rows, cols multiples of 32)
__global__ __launch_bounds__(256) void cvt_transpose_f16(const float* __restrict__ src,
                                                         _Float16* __restrict__ dst,
                                                         int rows, int cols) {
    __shared__ float tile[32][33];
    const int bx = blockIdx.x * 32;  // col tile
    const int by = blockIdx.y * 32;  // row tile
    const int tx = threadIdx.x, ty = threadIdx.y;
#pragma unroll
    for (int j = 0; j < 32; j += 8)
        tile[ty + j][tx] = src[(size_t)(by + ty + j) * cols + bx + tx];
    __syncthreads();
#pragma unroll
    for (int j = 0; j < 32; j += 8)
        dst[(size_t)(bx + ty + j) * rows + by + tx] = (_Float16)tile[tx][ty + j];
}

// ---------------- WMMA GEMM: C[M][N] f32 = A[M][K] f16 * BT[N][K]^T ----------------
// block = 256 threads = 8 waves, waves arranged 2(M) x 4(N); wave tile 32x64.
__global__ __launch_bounds__(256) void gemm_f16_wmma(const _Float16* __restrict__ A,
                                                     const _Float16* __restrict__ BT,
                                                     float* __restrict__ C,
                                                     int M, int N, int K) {
    const int lane = threadIdx.x & 31;
    const int wave = threadIdx.x >> 5;
    const int row0 = blockIdx.x * 64  + (wave >> 2) * 32;
    const int col0 = blockIdx.y * 256 + (wave & 3) * 64;

    const v8f vz = {0.f, 0.f, 0.f, 0.f, 0.f, 0.f, 0.f, 0.f};
    v8f acc[2][4];
#pragma unroll
    for (int i = 0; i < 2; ++i)
#pragma unroll
        for (int j = 0; j < 4; ++j) acc[i][j] = vz;

    for (int k = 0; k < K; k += 32) {
        v16h a0 = load_fragA(A, row0,      K, k, lane);
        v16h a1 = load_fragA(A, row0 + 16, K, k, lane);
#pragma unroll
        for (int j = 0; j < 4; ++j) {
            v16h b = load_fragB(BT, col0 + j * 16, K, k, lane);
            acc[0][j] = wmma_f16(a0, b, acc[0][j]);
            acc[1][j] = wmma_f16(a1, b, acc[1][j]);
        }
    }

    const int hi = lane >> 4, cn = lane & 15;
#pragma unroll
    for (int i = 0; i < 2; ++i)
#pragma unroll
        for (int j = 0; j < 4; ++j) {
            float* p = C + (size_t)(row0 + i * 16 + hi * 8) * N + col0 + j * 16 + cn;
#pragma unroll
            for (int v = 0; v < 8; ++v) p[(size_t)v * N] = acc[i][j][v];
        }
}

// ---------------- RoPE + split + layout kernel ----------------
// qkv f32 [SEQ][6144] -> q f16 [32][SEQ][128], k f16 [8][SEQ][128] (x0.5, roped),
//                        vT f16 [8][128][SEQ]
__global__ void rope_split_kernel(const int* __restrict__ positions,
                                  const float* __restrict__ qkv,
                                  _Float16* __restrict__ q,
                                  _Float16* __restrict__ k,
                                  _Float16* __restrict__ vT) {
    const int t = blockIdx.x;
    const float pos = (float)positions[t];
    const float* row = qkv + (size_t)t * QKV_N;

    // Q heads: 32 * 64 pairs
    for (int idx = threadIdx.x; idx < N_HEADS * 64; idx += blockDim.x) {
        const int h = idx >> 6, i = idx & 63;
        const float x1 = row[h * HD + i];
        const float x2 = row[h * HD + i + 64];
        const float fr = pos * __expf(NEG_LN_THETA_OVER_64 * (float)i);
        float s, c; __sincosf(fr, &s, &c);
        _Float16* qh = q + ((size_t)h * SEQ + t) * HD;
        qh[i]      = (_Float16)(x1 * c - x2 * s);
        qh[i + 64] = (_Float16)(x2 * c + x1 * s);
    }
    // K heads: 8 * 64 pairs (scaled by 0.5 — commutes with RoPE)
    for (int idx = threadIdx.x; idx < N_KVH * 64; idx += blockDim.x) {
        const int h = idx >> 6, i = idx & 63;
        const float x1 = KEY_MULT * row[Q_SIZE + h * HD + i];
        const float x2 = KEY_MULT * row[Q_SIZE + h * HD + i + 64];
        const float fr = pos * __expf(NEG_LN_THETA_OVER_64 * (float)i);
        float s, c; __sincosf(fr, &s, &c);
        _Float16* kh = k + ((size_t)h * SEQ + t) * HD;
        kh[i]      = (_Float16)(x1 * c - x2 * s);
        kh[i + 64] = (_Float16)(x2 * c + x1 * s);
    }
    // V transpose: vT[h][d][t]
    for (int idx = threadIdx.x; idx < N_KVH * HD; idx += blockDim.x) {
        const int h = idx >> 7, d = idx & 127;
        vT[((size_t)h * HD + d) * SEQ + t] =
            (_Float16)row[Q_SIZE + KV_SIZE + h * HD + d];
    }
}

// ---------------- flash attention (causal, GQA 4:1) ----------------
// grid (SEQ/64, 32 heads), block 128 = 4 waves; each wave owns 16 query rows.
__global__ __launch_bounds__(128) void attn_kernel(const _Float16* __restrict__ q,
                                                   const _Float16* __restrict__ kx,
                                                   const _Float16* __restrict__ vT,
                                                   _Float16* __restrict__ attn) {
    __shared__ __align__(16) _Float16 pshare[4][16 * 32];
    const int lane = threadIdx.x & 31;
    const int wave = threadIdx.x >> 5;
    const int h    = blockIdx.y;
    const int kvh  = h >> 2;                         // 32/8 = 4 Q heads per KV head
    const int qBase = blockIdx.x * 64 + wave * 16;
    const int hi = lane >> 4, cn = lane & 15;

    const _Float16* qh = q  + (size_t)h   * SEQ * HD;
    const _Float16* kh = kx + (size_t)kvh * SEQ * HD;
    const _Float16* vh = vT + (size_t)kvh * HD * SEQ;   // [128][SEQ]

    v16h qf[4];
#pragma unroll
    for (int dk = 0; dk < 4; ++dk) qf[dk] = load_fragA(qh, qBase, HD, dk * 32, lane);

    const v8f vz = {0.f, 0.f, 0.f, 0.f, 0.f, 0.f, 0.f, 0.f};
    v8f acc[8];
#pragma unroll
    for (int i = 0; i < 8; ++i) acc[i] = vz;
    float mrow[8], lrow[8];
#pragma unroll
    for (int v = 0; v < 8; ++v) { mrow[v] = -1e30f; lrow[v] = 0.f; }

    _Float16* my = &pshare[wave][0];

    for (int kv = 0; kv <= qBase + 15; kv += 32) {
        // S = Q * K^T  (16 x 32 scores as two 16x16 C fragments)
        v8f s0 = vz, s1 = vz;
#pragma unroll
        for (int dk = 0; dk < 4; ++dk) {
            v16h b0 = load_fragB(kh, kv,      HD, dk * 32, lane);
            v16h b1 = load_fragB(kh, kv + 16, HD, dk * 32, lane);
            s0 = wmma_f16(qf[dk], b0, s0);
            s1 = wmma_f16(qf[dk], b1, s1);
        }
        // online softmax (rows live in half-wave lane groups; reduce with shfl_xor 8..1)
#pragma unroll
        for (int v = 0; v < 8; ++v) {
            const int rowPos = qBase + hi * 8 + v;
            float a = s0[v] * ATTN_SCALE;
            float b = s1[v] * ATTN_SCALE;
            if (kv + cn      > rowPos) a = -1e30f;
            if (kv + 16 + cn > rowPos) b = -1e30f;
            float mx = fmaxf(a, b);
#pragma unroll
            for (int off = 8; off >= 1; off >>= 1) mx = fmaxf(mx, __shfl_xor(mx, off, 32));
            const float mnew  = fmaxf(mrow[v], mx);
            const float alpha = __expf(mrow[v] - mnew);
            const float pa = __expf(a - mnew);
            const float pb = __expf(b - mnew);
            float ps = pa + pb;
#pragma unroll
            for (int off = 8; off >= 1; off >>= 1) ps += __shfl_xor(ps, off, 32);
            lrow[v] = lrow[v] * alpha + ps;
            mrow[v] = mnew;
#pragma unroll
            for (int df = 0; df < 8; ++df) acc[df][v] *= alpha;
            // bounce P through LDS to re-lay C-fragment -> A-fragment
            my[(hi * 8 + v) * 32 + cn]      = (_Float16)pa;
            my[(hi * 8 + v) * 32 + 16 + cn] = (_Float16)pb;
        }
        asm volatile("s_wait_dscnt 0" ::: "memory");
        const _Float16* lp = my + cn * 32 + hi * 8;
        v8h plo = *(const v8h*)(lp);
        v8h phi = *(const v8h*)(lp + 16);
        v16h pf;
#pragma unroll
        for (int i = 0; i < 8; ++i) { pf[i] = plo[i]; pf[i + 8] = phi[i]; }
        // O += P * V   (B^T = V^T[d][kv], contiguous K)
#pragma unroll
        for (int df = 0; df < 8; ++df) {
            v16h vb = load_fragB(vh, df * 16, SEQ, kv, lane);
            acc[df] = wmma_f16(pf, vb, acc[df]);
        }
    }

    // epilogue: normalize, store f16 attn[SEQ][4096] at column h*128
#pragma unroll
    for (int df = 0; df < 8; ++df) {
        _Float16* op = attn + (size_t)(qBase + hi * 8) * Q_SIZE + h * HD + df * 16 + cn;
#pragma unroll
        for (int v = 0; v < 8; ++v)
            op[(size_t)v * Q_SIZE] = (_Float16)(acc[df][v] / lrow[v]);
    }
}

// ---------------- host-side orchestration ----------------
extern "C" void kernel_launch(void* const* d_in, const int* in_sizes, int n_in,
                              void* d_out, int out_size, void* d_ws, size_t ws_size,
                              hipStream_t stream) {
    const int*   positions = (const int*)d_in[0];
    const float* hidden    = (const float*)d_in[1];
    const float* w_qkv     = (const float*)d_in[2];
    const float* w_o       = (const float*)d_in[3];
    float*       out       = (float*)d_out;

    char* w = (char*)d_ws;
    const size_t off_hs   = 0;                                   // SEQ*HIDDEN f16
    const size_t off_wqkv = off_hs   + (size_t)SEQ * HIDDEN * 2; // QKV_N*HIDDEN f16 (transposed)
    const size_t off_wo   = off_wqkv + (size_t)QKV_N * HIDDEN * 2; // HIDDEN*HIDDEN f16 (transposed)
    const size_t off_qkv  = off_wo   + (size_t)HIDDEN * HIDDEN * 2; // SEQ*QKV_N f32 (reused by attn f16)
    const size_t off_q    = off_qkv  + (size_t)SEQ * QKV_N * 4;
    const size_t off_k    = off_q    + (size_t)SEQ * Q_SIZE * 2;
    const size_t off_vT   = off_k    + (size_t)SEQ * KV_SIZE * 2;

    _Float16* hs_f16  = (_Float16*)(w + off_hs);
    _Float16* wqkvT   = (_Float16*)(w + off_wqkv);
    _Float16* woT     = (_Float16*)(w + off_wo);
    float*    qkv_f32 = (float*)   (w + off_qkv);
    _Float16* attn    = (_Float16*)(w + off_qkv);   // reuse after rope consumed qkv
    _Float16* q_f16   = (_Float16*)(w + off_q);
    _Float16* k_f16   = (_Float16*)(w + off_k);
    _Float16* vT_f16  = (_Float16*)(w + off_vT);

    // 1) convert hidden_states, transpose-convert both weight matrices to f16
    cvt_f32_to_f16<<<2048, 256, 0, stream>>>(hidden, hs_f16, SEQ * HIDDEN);
    cvt_transpose_f16<<<dim3(QKV_N / 32, HIDDEN / 32), dim3(32, 8), 0, stream>>>(
        w_qkv, wqkvT, HIDDEN, QKV_N);
    cvt_transpose_f16<<<dim3(HIDDEN / 32, HIDDEN / 32), dim3(32, 8), 0, stream>>>(
        w_o, woT, HIDDEN, HIDDEN);

    // 2) QKV projection: [SEQ][QKV_N] f32 = hs_f16 @ wqkvT^T
    gemm_f16_wmma<<<dim3(SEQ / 64, QKV_N / 256), 256, 0, stream>>>(
        hs_f16, wqkvT, qkv_f32, SEQ, QKV_N, HIDDEN);

    // 3) RoPE + key scaling + head-major split (+ V transpose)
    rope_split_kernel<<<SEQ, 256, 0, stream>>>(positions, qkv_f32, q_f16, k_f16, vT_f16);

    // 4) causal GQA flash attention -> attn f16 [SEQ][4096]
    attn_kernel<<<dim3(SEQ / 64, N_HEADS), 128, 0, stream>>>(q_f16, k_f16, vT_f16, attn);

    // 5) output projection: d_out f32 = attn @ woT^T
    gemm_f16_wmma<<<dim3(SEQ / 64, HIDDEN / 256), 256, 0, stream>>>(
        attn, woT, out, SEQ, HIDDEN, HIDDEN);
}
